// EnhancedGatedFusion_71889162600548
// MI455X (gfx1250) — matrix operand order repaired and checked by reading, not compiled, not run on hardware
//
#include <hip/hip_runtime.h>
#include <hip/hip_bf16.h>

typedef __attribute__((ext_vector_type(4)))  float    v4f;
typedef __attribute__((ext_vector_type(8)))  float    v8f;
typedef __attribute__((ext_vector_type(8)))  _Float16 v8h;
typedef __attribute__((ext_vector_type(16))) _Float16 v16h;

#define NROWS 16384
#define DDIM  2048
#define KDIM  4096

#define TM   32     // rows per workgroup tile
#define TND  256    // d-columns per workgroup tile
#define BK   32     // K chunk (matches 16x16x32 f16 WMMA)
#define KP   40     // padded LDS row length in halves (32 + 8 pad -> kills bank conflicts)

#define LN_EPS 1e-5f

// ---------------------------------------------------------------------------
// CDNA5 async global->LDS staging (ASYNCcnt-tracked, no VGPR round trip).
// LDS address operand is the wave-relative LDS byte offset; per ISA 10.2 the
// low 32 bits of a flat shared-aperture address are exactly that offset.
// ---------------------------------------------------------------------------
__device__ __forceinline__ void async_load_b128(void* lds_ptr, const void* gptr)
{
    uint32_t lo = (uint32_t)(uintptr_t)lds_ptr;
    asm volatile("global_load_async_to_lds_b128 %0, %1, off"
                 :: "v"(lo), "v"(gptr)
                 : "memory");
}

__device__ __forceinline__ void wait_async0()
{
    asm volatile("s_wait_asynccnt 0x0" ::: "memory");
}

// ---------------------------------------------------------------------------
// f32 -> f16 conversion (count must be a multiple of 8; all arrays here are)
// ---------------------------------------------------------------------------
__global__ void cvt_f32_to_f16_kernel(const float* __restrict__ src,
                                      _Float16* __restrict__ dst,
                                      long long n8)
{
    long long i = (long long)blockIdx.x * blockDim.x + threadIdx.x;
    if (i >= n8) return;
    const v4f* s = reinterpret_cast<const v4f*>(src) + 2 * i;
    v4f a = s[0], b = s[1];
    v8h o;
    o[0] = (_Float16)a[0]; o[1] = (_Float16)a[1];
    o[2] = (_Float16)a[2]; o[3] = (_Float16)a[3];
    o[4] = (_Float16)b[0]; o[5] = (_Float16)b[1];
    o[6] = (_Float16)b[2]; o[7] = (_Float16)b[3];
    reinterpret_cast<v8h*>(dst)[i] = o;
}

// ---------------------------------------------------------------------------
// concat(emb1, emb2) -> f16 [NROWS][KDIM]
// ---------------------------------------------------------------------------
__global__ void concat_cvt_kernel(const float* __restrict__ e1,
                                  const float* __restrict__ e2,
                                  _Float16* __restrict__ Ah)
{
    long long id = (long long)blockIdx.x * blockDim.x + threadIdx.x; // one per 8 elems of D
    long long total = (long long)NROWS * (DDIM / 8);
    if (id >= total) return;
    long long n  = id / (DDIM / 8);
    int       c8 = (int)(id % (DDIM / 8));

    const v4f* s1 = reinterpret_cast<const v4f*>(e1 + n * DDIM + c8 * 8);
    const v4f* s2 = reinterpret_cast<const v4f*>(e2 + n * DDIM + c8 * 8);
    v4f a = s1[0], b = s1[1], c = s2[0], d = s2[1];

    v8h o1, o2;
    o1[0]=(_Float16)a[0]; o1[1]=(_Float16)a[1]; o1[2]=(_Float16)a[2]; o1[3]=(_Float16)a[3];
    o1[4]=(_Float16)b[0]; o1[5]=(_Float16)b[1]; o1[6]=(_Float16)b[2]; o1[7]=(_Float16)b[3];
    o2[0]=(_Float16)c[0]; o2[1]=(_Float16)c[1]; o2[2]=(_Float16)c[2]; o2[3]=(_Float16)c[3];
    o2[4]=(_Float16)d[0]; o2[5]=(_Float16)d[1]; o2[6]=(_Float16)d[2]; o2[7]=(_Float16)d[3];

    _Float16* row = Ah + n * KDIM;
    *reinterpret_cast<v8h*>(row + c8 * 8)        = o1;   // first half  = emb1
    *reinterpret_cast<v8h*>(row + DDIM + c8 * 8) = o2;   // second half = emb2
}

// ---------------------------------------------------------------------------
// Fused gate GEMM.
//  Wh layout: [10][DDIM][KDIM] f16, matrices 0..2 = Wf(g), 3..5 = Wi(g),
//  6..8 = Wu(g), 9 = Wo.
//  Pass g in {0,1,2}: acc {Wf_g, Wi_g, Wu_g} together; pass 3: Wo.
//  Writes x = emb1 + sigma(o) * tanh(fused) to xout (pre-LayerNorm).
// ---------------------------------------------------------------------------
__global__ __launch_bounds__(256)
void fused_gate_gemm_kernel(const _Float16* __restrict__ Ah,
                            const _Float16* __restrict__ Wh,
                            const float* __restrict__ emb1,
                            const float* __restrict__ bf,
                            const float* __restrict__ bi,
                            const float* __restrict__ bu,
                            const float* __restrict__ bo,
                            const float* __restrict__ attw,
                            float* __restrict__ xout)
{
    __shared__ _Float16 Ach[2][TM][KP];        //  2 x  2.5 KB
    __shared__ _Float16 Bch[2][3][TND][KP];    //  2 x 60   KB

    const int tid   = threadIdx.x;
    const int lane  = tid & 31;
    const int wv    = tid >> 5;
    const int rb    = wv & 1;        // row sub-block (0..1)
    const int cq    = wv >> 1;       // column quad   (0..3), 4 col-blocks each
    const int nbase = blockIdx.y * TM;
    const int dbase = blockIdx.x * TND;

    // softmax over att_w (G=3), uniform per thread
    float a0 = attw[0], a1 = attw[1], a2 = attw[2];
    float mx = fmaxf(a0, fmaxf(a1, a2));
    float e0 = expf(a0 - mx), e1 = expf(a1 - mx), e2 = expf(a2 - mx);
    float inv = 1.0f / (e0 + e1 + e2);
    float wgt[3] = { e0 * inv, e1 * inv, e2 * inv };

    const int halfsel = lane >> 4;   // 0: lanes 0-15, 1: lanes 16-31
    const int nn      = lane & 15;

    // ISA 7.12.2 fragment addressing (16-bit, 16x16x32):
    //  A: lane<16 -> K {0..7,16..23};  lane>=16 -> K {8..15,24..31}
    //  B: lane<16 -> K {0..7,16..23};  lane>=16 -> K {8..15,24..31}
    const int a_row = rb * 16 + nn;
    const int a_k0  = halfsel * 8;
    const int b_k0  = halfsel * 8;

    v8f fused[4];
    #pragma unroll
    for (int i = 0; i < 4; ++i)
        #pragma unroll
        for (int r = 0; r < 8; ++r) fused[i][r] = 0.0f;

    for (int pass = 0; pass < 4; ++pass) {
        const int J = (pass < 3) ? 3 : 1;

        v8f acc[3][4];
        #pragma unroll
        for (int j = 0; j < 3; ++j)
            #pragma unroll
            for (int i = 0; i < 4; ++i)
                #pragma unroll
                for (int r = 0; r < 8; ++r) acc[j][i][r] = 0.0f;

        long long mb[3];
        if (pass < 3) {
            mb[0] = (long long)(0 + pass) * DDIM * KDIM;  // Wf_g
            mb[1] = (long long)(3 + pass) * DDIM * KDIM;  // Wi_g
            mb[2] = (long long)(6 + pass) * DDIM * KDIM;  // Wu_g
        } else {
            mb[0] = (long long)9 * DDIM * KDIM;           // Wo
            mb[1] = mb[0]; mb[2] = mb[0];
        }

        // Issue async global->LDS staging for one K chunk into buffer `buf`.
        auto stage = [&](int buf, int k0) {
            if (tid < 128) {                               // A: 128 x 16B chunks
                int r = tid >> 2, c = tid & 3;
                async_load_b128(&Ach[buf][r][c * 8],
                                Ah + (long long)(nbase + r) * KDIM + k0 + c * 8);
            }
            for (int it2 = 0; it2 < J * 4; ++it2) {        // B: J x 1024 x 16B chunks
                int id  = tid + it2 * 256;
                int j   = id >> 10;
                int rem = id & 1023;
                int d   = rem >> 2, c = rem & 3;
                async_load_b128(&Bch[buf][j][d][c * 8],
                                Wh + mb[j] + (long long)(dbase + d) * KDIM + k0 + c * 8);
            }
        };

        const int NK = KDIM / BK;
        stage(0, 0);
        for (int it = 0; it < NK; ++it) {
            const int b = it & 1;
            wait_async0();          // my async stores into buffer b landed in LDS
            __syncthreads();        // everyone's did; prev reads of b^1 finished
            if (it + 1 < NK) stage(b ^ 1, (it + 1) * BK);

            // A fragment (shared by all J matrices and 4 col-blocks)
            v8h alo = *reinterpret_cast<const v8h*>(&Ach[b][a_row][a_k0]);
            v8h ahi = *reinterpret_cast<const v8h*>(&Ach[b][a_row][a_k0 + 16]);
            v16h afrag = __builtin_shufflevector(alo, ahi,
                0,1,2,3,4,5,6,7,8,9,10,11,12,13,14,15);

            #pragma unroll
            for (int j = 0; j < 3; ++j) {
                if (j >= J) break;   // uniform branch, EXEC stays all-ones
                // batch all 4 fragment loads, then 4 WMMAs (hide LDS latency)
                v8h bl[4], bh[4];
                #pragma unroll
                for (int i = 0; i < 4; ++i) {
                    int brow = cq * 64 + i * 16 + nn;
                    bl[i] = *reinterpret_cast<const v8h*>(&Bch[b][j][brow][b_k0]);
                    bh[i] = *reinterpret_cast<const v8h*>(&Bch[b][j][brow][b_k0 + 16]);
                }
                #pragma unroll
                for (int i = 0; i < 4; ++i) {
                    v16h bfrag = __builtin_shufflevector(bl[i], bh[i],
                        0,1,2,3,4,5,6,7,8,9,10,11,12,13,14,15);
                    acc[j][i] = __builtin_amdgcn_wmma_f32_16x16x32_f16(
                        false, afrag, false, bfrag, (short)0, acc[j][i],
                        false, false);
                }
            }
        }

        // ---- epilogue ----
        if (pass < 3) {
            const float wg = wgt[pass];
            #pragma unroll
            for (int i = 0; i < 4; ++i) {
                int d = dbase + cq * 64 + i * 16 + nn;
                float bfv = bf[pass * DDIM + d];
                float biv = bi[pass * DDIM + d];
                float buv = bu[pass * DDIM + d];
                #pragma unroll
                for (int r = 0; r < 8; ++r) {
                    int row = nbase + rb * 16 + r + halfsel * 8;
                    float f  = 1.0f / (1.0f + expf(-(acc[0][i][r] + bfv)));
                    float ig = 1.0f / (1.0f + expf(-(acc[1][i][r] + biv)));
                    float u  = tanhf(acc[2][i][r] + buv);
                    float e1v = emb1[(long long)row * DDIM + d];
                    fused[i][r] += wg * (f * e1v + ig * u);
                }
            }
        } else {
            #pragma unroll
            for (int i = 0; i < 4; ++i) {
                int d = dbase + cq * 64 + i * 16 + nn;
                float bov = bo[d];
                #pragma unroll
                for (int r = 0; r < 8; ++r) {
                    int row = nbase + rb * 16 + r + halfsel * 8;
                    float o   = 1.0f / (1.0f + expf(-(acc[0][i][r] + bov)));
                    float e1v = emb1[(long long)row * DDIM + d];
                    xout[(long long)row * DDIM + d] = e1v + o * tanhf(fused[i][r]);
                }
            }
        }
    }
}

// ---------------------------------------------------------------------------
// In-place LayerNorm over rows of x [NROWS][DDIM]
// ---------------------------------------------------------------------------
__global__ void ln_kernel(float* __restrict__ x,
                          const float* __restrict__ gamma,
                          const float* __restrict__ beta)
{
    __shared__ float red[256];
    const int row = blockIdx.x;
    const int tid = threadIdx.x;

    float v[8];
    float s = 0.0f;
    #pragma unroll
    for (int i = 0; i < 8; ++i) {
        v[i] = x[(long long)row * DDIM + tid + i * 256];
        s += v[i];
    }
    red[tid] = s;
    __syncthreads();
    for (int off = 128; off > 0; off >>= 1) {
        if (tid < off) red[tid] += red[tid + off];
        __syncthreads();
    }
    const float mu = red[0] / (float)DDIM;
    __syncthreads();

    float q = 0.0f;
    #pragma unroll
    for (int i = 0; i < 8; ++i) { float c = v[i] - mu; q += c * c; }
    red[tid] = q;
    __syncthreads();
    for (int off = 128; off > 0; off >>= 1) {
        if (tid < off) red[tid] += red[tid + off];
        __syncthreads();
    }
    const float var  = red[0] / (float)DDIM;
    const float rstd = rsqrtf(var + LN_EPS);

    #pragma unroll
    for (int i = 0; i < 8; ++i) {
        int d = tid + i * 256;
        x[(long long)row * DDIM + d] = (v[i] - mu) * rstd * gamma[d] + beta[d];
    }
}

// ---------------------------------------------------------------------------
extern "C" void kernel_launch(void* const* d_in, const int* in_sizes, int n_in,
                              void* d_out, int out_size, void* d_ws, size_t ws_size,
                              hipStream_t stream)
{
    const float* emb1  = (const float*)d_in[0];
    const float* emb2  = (const float*)d_in[1];
    const float* Wf    = (const float*)d_in[2];
    const float* bf    = (const float*)d_in[3];
    const float* Wi    = (const float*)d_in[4];
    const float* bi    = (const float*)d_in[5];
    const float* Wu    = (const float*)d_in[6];
    const float* bu    = (const float*)d_in[7];
    const float* Wo    = (const float*)d_in[8];
    const float* bo    = (const float*)d_in[9];
    const float* attw  = (const float*)d_in[10];
    const float* gamma = (const float*)d_in[11];
    const float* beta  = (const float*)d_in[12];
    float* out = (float*)d_out;

    // workspace: Ah f16 [NROWS][KDIM] then Wh f16 [10][DDIM][KDIM]  (~302 MB)
    _Float16* Ah = (_Float16*)d_ws;
    _Float16* Wh = Ah + (size_t)NROWS * KDIM;
    const long long MAT = (long long)DDIM * KDIM;  // elems per weight matrix

    // 1) concat + convert activations
    {
        long long total = (long long)NROWS * (DDIM / 8);
        int blocks = (int)((total + 255) / 256);
        concat_cvt_kernel<<<blocks, 256, 0, stream>>>(emb1, emb2, Ah);
    }
    // 2) convert weights into the [10] arena
    {
        long long n8f = 3 * MAT / 8;                 // Wf, Wi, Wu each 3 matrices
        int blkf = (int)((n8f + 255) / 256);
        cvt_f32_to_f16_kernel<<<blkf, 256, 0, stream>>>(Wf, Wh + 0 * MAT, n8f);
        cvt_f32_to_f16_kernel<<<blkf, 256, 0, stream>>>(Wi, Wh + 3 * MAT, n8f);
        cvt_f32_to_f16_kernel<<<blkf, 256, 0, stream>>>(Wu, Wh + 6 * MAT, n8f);
        long long n8o = MAT / 8;
        int blko = (int)((n8o + 255) / 256);
        cvt_f32_to_f16_kernel<<<blko, 256, 0, stream>>>(Wo, Wh + 9 * MAT, n8o);
    }
    // 3) fused gate GEMM -> x (pre-LN) in d_out
    {
        dim3 grid(DDIM / TND, NROWS / TM);   // (8, 512): d-tiles fastest => weights stay L2-hot
        fused_gate_gemm_kernel<<<grid, 256, 0, stream>>>(
            Ah, Wh, emb1, bf, bi, bu, bo, attw, out);
    }
    // 4) LayerNorm in place
    ln_kernel<<<NROWS, 256, 0, stream>>>(out, gamma, beta);

    (void)in_sizes; (void)n_in; (void)out_size; (void)ws_size;
}
